// TemporalRefinementNetwork_3272765079754
// MI455X (gfx1250) — matrix operand order repaired and checked by reading, not compiled, not run on hardware
//
#include <hip/hip_runtime.h>
#include <math.h>

typedef _Float16 half8  __attribute__((ext_vector_type(8)));
typedef _Float16 half16 __attribute__((ext_vector_type(16)));
typedef float    f32x8  __attribute__((ext_vector_type(8)));

#define QN     256
#define TN     64
#define CF     64
#define HFM    128
#define WFM    128
#define NROWS  (QN * TN)      /* 16384 */
#define HID    128
#define P49    49
#define INDIM  276
#define KPAD   288
#define FSTRIDE 4.0f

__device__ __forceinline__ float gelu_exact(float x) {
    return 0.5f * x * (1.0f + erff(x * 0.70710678118654752440f));
}

// ---------------------------------------------------------------------------
// Weight conversion to f16 (done every launch; deterministic, cheap)
// ---------------------------------------------------------------------------
__global__ void cvt_f16_kernel(const float* __restrict__ s, _Float16* __restrict__ d, int n) {
    int i = blockIdx.x * blockDim.x + threadIdx.x;
    if (i < n) d[i] = (_Float16)s[i];
}

__global__ void cvt_w_in_kernel(const float* __restrict__ s, _Float16* __restrict__ d) {
    int i = blockIdx.x * blockDim.x + threadIdx.x;
    if (i >= KPAD * HID) return;
    int r = i / HID, c = i % HID;
    d[i] = (r < INDIM) ? (_Float16)s[r * HID + c] : (_Float16)0.0f;
}

// ---------------------------------------------------------------------------
// Correlation + soft-argmax delta + Fourier PE + input assembly (f16 row of 288)
// One 64-thread block per (q,t); thread = feature channel.
// ---------------------------------------------------------------------------
__global__ __launch_bounds__(64)
void corr_assemble_kernel(const float* __restrict__ qf,
                          const float* __restrict__ fmaps,
                          float* __restrict__ coords,
                          const float* __restrict__ hidden,
                          _Float16* __restrict__ X)
{
    const int n   = blockIdx.x;
    const int q   = n / TN;
    const int t   = n % TN;
    const int tid = threadIdx.x;          // channel index 0..63
    const int lane = tid & 31, wid = tid >> 5;

    __shared__ float sPart[2][P49];
    __shared__ float sCorr[P49];
    __shared__ float sPos[2];

    const float cx = coords[2 * n + 0];
    const float cy = coords[2 * n + 1];
    const float fx = cx * (1.0f / FSTRIDE);
    const float fy = cy * (1.0f / FSTRIDE);
    const float qv = qf[q * CF + tid];
    const float* fm = fmaps + ((size_t)t * CF + tid) * (size_t)(HFM * WFM);

    for (int p = 0; p < P49; ++p) {
        const float ox = (float)(p % 7 - 3), oy = (float)(p / 7 - 3);
        const float sx = fx + ox, sy = fy + oy;
        const float x0f = floorf(sx), y0f = floorf(sy);
        const float wx = sx - x0f, wy = sy - y0f;
        int x0 = (int)x0f, y0 = (int)y0f;
        int x1 = x0 + 1, y1 = y0 + 1;
        x0 = min(max(x0, 0), WFM - 1); x1 = min(max(x1, 0), WFM - 1);
        y0 = min(max(y0, 0), HFM - 1); y1 = min(max(y1, 0), HFM - 1);
        const float v00 = fm[y0 * WFM + x0], v01 = fm[y0 * WFM + x1];
        const float v10 = fm[y1 * WFM + x0], v11 = fm[y1 * WFM + x1];
        const float s = (v00 * (1.f - wx) + v01 * wx) * (1.f - wy)
                      + (v10 * (1.f - wx) + v11 * wx) * wy;
        float red = s * qv;
        red += __shfl_xor(red, 16, 32);
        red += __shfl_xor(red, 8, 32);
        red += __shfl_xor(red, 4, 32);
        red += __shfl_xor(red, 2, 32);
        red += __shfl_xor(red, 1, 32);
        if (lane == 0) sPart[wid][p] = red;
    }
    __syncthreads();
    if (tid < P49) sCorr[tid] = sPart[0][tid] + sPart[1][tid];
    __syncthreads();

    if (tid == 0) {
        float mx = -1e30f;
        for (int p = 0; p < P49; ++p) mx = fmaxf(mx, sCorr[p]);
        float sum = 0.f;
        for (int p = 0; p < P49; ++p) sum += expf((sCorr[p] - mx) * 10.0f);
        const float inv = 1.0f / sum;
        float sax = 0.f, say = 0.f;
        for (int p = 0; p < P49; ++p) {
            const float w = expf((sCorr[p] - mx) * 10.0f) * inv;
            sax += w * (float)(p % 7 - 3) * FSTRIDE;
            say += w * (float)(p / 7 - 3) * FSTRIDE;
        }
        const float nx = cx + sax, ny = cy + say;
        coords[2 * n + 0] = nx; coords[2 * n + 1] = ny;
        sPos[0] = nx; sPos[1] = ny;
    }
    __syncthreads();

    _Float16* Xr = X + (size_t)n * KPAD;
    if (tid < P49) Xr[tid] = (_Float16)sCorr[tid];

    const float pv0 = sPos[0] * (1.0f / 512.0f);
    const float pv1 = sPos[1] * (1.0f / 512.0f);
    const float pv2 = (float)t * (1.0f / 63.0f);
    for (int j = tid; j < 99; j += 64) {
        float val;
        if (j < 3) {
            val = (j == 0) ? pv0 : ((j == 1) ? pv1 : pv2);
        } else {
            const int u = j - 3;
            const int dim = u >> 5;          // 0..2
            const int k = u & 31;            // 0..31
            const int band = k & 15;
            const float freq = 3.14159265358979323846f * exp2f((float)band * (9.0f / 15.0f));
            const float pvv = (dim == 0) ? pv0 : ((dim == 1) ? pv1 : pv2);
            const float a = pvv * freq;
            val = (k < 16) ? sinf(a) : cosf(a);
        }
        Xr[P49 + j] = (_Float16)val;
    }
    for (int d2 = tid; d2 < HID; d2 += 64)
        Xr[P49 + 99 + d2] = (_Float16)hidden[(size_t)n * HID + d2];
    if (tid < KPAD - INDIM) Xr[INDIM + tid] = (_Float16)0.0f;
}

// ---------------------------------------------------------------------------
// WMMA GEMM: C[M,N] = A[M,K] * B[K,N] (+bias, optional gelu->f16 / residual)
// 256 threads, 8 waves; wave -> 32x32 tile (2x2 WMMA); WG tile 128x64; K-step 32.
// EPI: 0 = bias -> f32 out, 1 = bias+gelu -> f16 out, 2 = bias+residual -> f32 out
// ---------------------------------------------------------------------------
__device__ __forceinline__ half16 frag_from(half8 lo, half8 hi) {
    return __builtin_shufflevector(lo, hi, 0,1,2,3,4,5,6,7,8,9,10,11,12,13,14,15);
}

template <int EPI>
__global__ __launch_bounds__(256)
void gemm_wmma_kernel(const _Float16* __restrict__ A, const _Float16* __restrict__ B,
                      const float* __restrict__ bias, const float* __restrict__ res,
                      float* __restrict__ Cf, _Float16* __restrict__ Ch,
                      int K, int N)
{
    __shared__ __align__(16) _Float16 As[128 * 40];   // padded stride 40 halves
    __shared__ __align__(16) _Float16 Bs[64 * 40];    // stored K-major per column

    const int rowBase = blockIdx.x * 128;
    const int colBase = blockIdx.y * 64;
    const int tid  = threadIdx.x;
    const int lane = tid & 31;
    const int wv   = tid >> 5;
    const int waveM = wv & 3;      // 0..3 -> 32-row slice
    const int waveN = wv >> 2;     // 0..1 -> 32-col slice

    f32x8 acc00 = {0.f,0.f,0.f,0.f,0.f,0.f,0.f,0.f};
    f32x8 acc01 = acc00, acc10 = acc00, acc11 = acc00;

    const int nK = K >> 5;
    for (int kt = 0; kt < nK; ++kt) {
        const int k0 = kt << 5;
        // stage A tile: 128 rows x 32 halves (uint4 = 8 halves)
        for (int i = tid; i < 512; i += 256) {
            const int r = i >> 2, c = i & 3;
            const uint4 v = *(const uint4*)(A + (size_t)(rowBase + r) * K + k0 + c * 8);
            *(uint4*)(As + r * 40 + c * 8) = v;
        }
        // stage B tile transposed: Bs[n][k], 64 cols x 32 rows
        {
            const int r = tid >> 3, c = tid & 7;
            const uint4 v = *(const uint4*)(B + (size_t)(k0 + r) * N + colBase + c * 8);
            const _Float16* hv = (const _Float16*)&v;
#pragma unroll
            for (int j = 0; j < 8; ++j) Bs[(c * 8 + j) * 40 + r] = hv[j];
        }
        // prefetch next K-tile into cache while this tile computes
        // (lowers to global_prefetch_b8; memory-bound workload -> hide latency)
        if (kt + 1 < nK) {
            const int k0n = k0 + 32;
            if (tid < 128)
                __builtin_prefetch(A + (size_t)(rowBase + tid) * K + k0n, 0, 3);
            else if (tid < 160)
                __builtin_prefetch(B + (size_t)(k0n + (tid - 128)) * N + colBase, 0, 3);
        }
        __syncthreads();

        // fragment gathers per ISA 16-bit A 16x32 / B 32x16 layouts
        const int m   = lane & 15;
        const int hiA = (lane >> 4) << 3;          // 0 or 8 halves
        const _Float16* ra0 = As + (waveM * 32 + m) * 40;
        const _Float16* ra1 = As + (waveM * 32 + 16 + m) * 40;
        const half16 a0 = frag_from(*(const half8*)(ra0 + hiA), *(const half8*)(ra0 + 16 + hiA));
        const half16 a1 = frag_from(*(const half8*)(ra1 + hiA), *(const half8*)(ra1 + 16 + hiA));
        const int kbB = (lane >> 4) << 3;          // 0 or 8
        const _Float16* rb0 = Bs + (waveN * 32 + m) * 40 + kbB;
        const _Float16* rb1 = Bs + (waveN * 32 + 16 + m) * 40 + kbB;
        const half16 b0 = frag_from(*(const half8*)(rb0), *(const half8*)(rb0 + 16));
        const half16 b1 = frag_from(*(const half8*)(rb1), *(const half8*)(rb1 + 16));

        acc00 = __builtin_amdgcn_wmma_f32_16x16x32_f16(false, a0, false, b0, (short)0, acc00, false, false);
        acc01 = __builtin_amdgcn_wmma_f32_16x16x32_f16(false, a0, false, b1, (short)0, acc01, false, false);
        acc10 = __builtin_amdgcn_wmma_f32_16x16x32_f16(false, a1, false, b0, (short)0, acc10, false, false);
        acc11 = __builtin_amdgcn_wmma_f32_16x16x32_f16(false, a1, false, b1, (short)0, acc11, false, false);
        __syncthreads();
    }

    // epilogue (C/D 16x16 f32 layout: lane<16 -> M=r, lane>=16 -> M=8+r; N=lane&15)
#pragma unroll
    for (int sm = 0; sm < 2; ++sm) {
#pragma unroll
        for (int sn = 0; sn < 2; ++sn) {
            const f32x8 accv = (sm == 0) ? (sn == 0 ? acc00 : acc01)
                                         : (sn == 0 ? acc10 : acc11);
            const int cc    = colBase + waveN * 32 + sn * 16 + (lane & 15);
            const int rbase = rowBase + waveM * 32 + sm * 16 + ((lane >> 4) << 3);
            const float bv  = bias[cc];
#pragma unroll
            for (int r = 0; r < 8; ++r) {
                const int rr = rbase + r;
                float v = accv[r] + bv;
                if (EPI == 2) v += res[(size_t)rr * N + cc];
                if (EPI == 1) Ch[(size_t)rr * N + cc] = (_Float16)gelu_exact(v);
                else          Cf[(size_t)rr * N + cc] = v;
            }
        }
    }
}

// ---------------------------------------------------------------------------
// Fused LN1 -> depthwise conv1d(T, k=5, zero-pad) -> residual -> LN2 -> f16
// One 256-thread block per (b,q) strip of T=64 x D=128, LDS-resident.
// Params are pre-offset by block index at launch.
// ---------------------------------------------------------------------------
__global__ __launch_bounds__(256)
void block_pre_kernel(float* __restrict__ h, _Float16* __restrict__ z2,
                      const float* __restrict__ ln1_g, const float* __restrict__ ln1_b,
                      const float* __restrict__ conv_w, const float* __restrict__ conv_b,
                      const float* __restrict__ ln2_g, const float* __restrict__ ln2_b)
{
    __shared__ float sH[TN * HID];
    __shared__ float sSum[TN][4];
    __shared__ float sSq[TN][4];
    __shared__ float sMean[TN], sRstd[TN];
    __shared__ float sCw[HID * 5], sCb[HID];
    __shared__ float sG1[HID], sB1[HID], sG2[HID], sB2[HID];

    const int bq  = blockIdx.x;
    const int tid = threadIdx.x;
    const int t   = tid & 63;
    const int rep = tid >> 6;     // 0..3
    const int d0  = rep * 32;
    float* hg = h + (size_t)bq * TN * HID;

    for (int i = tid; i < TN * HID; i += 256) sH[i] = hg[i];
    for (int i = tid; i < HID * 5; i += 256)  sCw[i] = conv_w[i];
    if (tid < HID) {
        sCb[tid] = conv_b[tid];
        sG1[tid] = ln1_g[tid]; sB1[tid] = ln1_b[tid];
        sG2[tid] = ln2_g[tid]; sB2[tid] = ln2_b[tid];
    }
    __syncthreads();

    // LN1 statistics
    {
        float s = 0.f, s2 = 0.f;
#pragma unroll
        for (int j = 0; j < 32; ++j) {
            const float v = sH[t * HID + d0 + j];
            s += v; s2 += v * v;
        }
        sSum[t][rep] = s; sSq[t][rep] = s2;
    }
    __syncthreads();
    if (tid < TN) {
        const float s  = sSum[tid][0] + sSum[tid][1] + sSum[tid][2] + sSum[tid][3];
        const float s2 = sSq[tid][0] + sSq[tid][1] + sSq[tid][2] + sSq[tid][3];
        const float m  = s * (1.0f / HID);
        const float var = s2 * (1.0f / HID) - m * m;
        sMean[tid] = m; sRstd[tid] = rsqrtf(var + 1e-5f);
    }
    __syncthreads();

    // depthwise conv over t on LN1(x), + residual; accumulate LN2 stats
    float hn[32];
    float s = 0.f, s2 = 0.f;
#pragma unroll
    for (int j = 0; j < 32; ++j) {
        const int d = d0 + j;
        float acc = sCb[d];
#pragma unroll
        for (int k = 0; k < 5; ++k) {
            const int tt = t + k - 2;
            if (tt >= 0 && tt < TN) {
                const float ln1v = (sH[tt * HID + d] - sMean[tt]) * sRstd[tt] * sG1[d] + sB1[d];
                acc += ln1v * sCw[d * 5 + k];
            }
        }
        const float v = sH[t * HID + d] + acc;
        hn[j] = v; s += v; s2 += v * v;
    }
    __syncthreads();   // all reads of sH / sMean / sRstd done
    sSum[t][rep] = s; sSq[t][rep] = s2;
    __syncthreads();
    if (tid < TN) {
        const float ss  = sSum[tid][0] + sSum[tid][1] + sSum[tid][2] + sSum[tid][3];
        const float ss2 = sSq[tid][0] + sSq[tid][1] + sSq[tid][2] + sSq[tid][3];
        const float m   = ss * (1.0f / HID);
        const float var = ss2 * (1.0f / HID) - m * m;
        sMean[tid] = m; sRstd[tid] = rsqrtf(var + 1e-5f);
    }
    __syncthreads();

    const float m2 = sMean[t], r2 = sRstd[t];
    _Float16* zg = z2 + (size_t)bq * TN * HID;
#pragma unroll
    for (int j = 0; j < 32; ++j) {
        const int d = d0 + j;
        hg[t * HID + d] = hn[j];
        zg[t * HID + d] = (_Float16)((hn[j] - m2) * r2 * sG2[d] + sB2[d]);
    }
}

// ---------------------------------------------------------------------------
// coords += h @ w_delta + b_delta  (w_delta is [128,2])
// ---------------------------------------------------------------------------
__global__ void delta_update_kernel(float* __restrict__ coords, const float* __restrict__ h,
                                    const float* __restrict__ w_delta,
                                    const float* __restrict__ b_delta)
{
    const int n = blockIdx.x * blockDim.x + threadIdx.x;
    if (n >= NROWS) return;
    float dx = b_delta[0], dy = b_delta[1];
    const float* hr = h + (size_t)n * HID;
    for (int d = 0; d < HID; ++d) {
        const float v = hr[d];
        dx += v * w_delta[2 * d + 0];
        dy += v * w_delta[2 * d + 1];
    }
    coords[2 * n + 0] += dx;
    coords[2 * n + 1] += dy;
}

// ---------------------------------------------------------------------------
// vis head: gelu(hidden @ vis_w1 + b1) @ vis_w2 + b2; also copy coords to out.
// out layout: [coords 16384*2][vis 16384]
// ---------------------------------------------------------------------------
__global__ __launch_bounds__(64)
void finalize_kernel(const float* __restrict__ hidden, const float* __restrict__ coords,
                     const float* __restrict__ w1, const float* __restrict__ b1,
                     const float* __restrict__ w2, const float* __restrict__ b2,
                     float* __restrict__ out)
{
    const int n = blockIdx.x;
    const int u = threadIdx.x;              // hidden unit 0..63
    const int lane = u & 31, wid = u >> 5;
    __shared__ float sp[2];

    float a = b1[u];
    const float* hr = hidden + (size_t)n * HID;
    for (int d = 0; d < HID; ++d) a += hr[d] * w1[d * 64 + u];
    float contrib = gelu_exact(a) * w2[u];
    contrib += __shfl_xor(contrib, 16, 32);
    contrib += __shfl_xor(contrib, 8, 32);
    contrib += __shfl_xor(contrib, 4, 32);
    contrib += __shfl_xor(contrib, 2, 32);
    contrib += __shfl_xor(contrib, 1, 32);
    if (lane == 0) sp[wid] = contrib;
    __syncthreads();
    if (u == 0) out[2 * NROWS + n] = sp[0] + sp[1] + b2[0];
    if (u < 2)  out[2 * n + u] = coords[2 * n + u];
}

// ---------------------------------------------------------------------------
extern "C" void kernel_launch(void* const* d_in, const int* in_sizes, int n_in,
                              void* d_out, int out_size, void* d_ws, size_t ws_size,
                              hipStream_t stream)
{
    (void)in_sizes; (void)n_in; (void)out_size; (void)ws_size;

    const float* qf      = (const float*)d_in[0];
    const float* fmaps   = (const float*)d_in[1];
    const float* pos     = (const float*)d_in[2];
    /* d_in[3] = feature_stride (int scalar == 4, baked into constants) */
    const float* w_in    = (const float*)d_in[4];
    const float* b_in    = (const float*)d_in[5];
    const float* ln1_g   = (const float*)d_in[6];
    const float* ln1_b   = (const float*)d_in[7];
    const float* conv_w  = (const float*)d_in[8];
    const float* conv_b  = (const float*)d_in[9];
    const float* ln2_g   = (const float*)d_in[10];
    const float* ln2_b   = (const float*)d_in[11];
    const float* mlp_w1  = (const float*)d_in[12];
    const float* mlp_b1  = (const float*)d_in[13];
    const float* mlp_w2  = (const float*)d_in[14];
    const float* mlp_b2  = (const float*)d_in[15];
    const float* w_delta = (const float*)d_in[16];
    const float* b_delta = (const float*)d_in[17];
    const float* vis_w1  = (const float*)d_in[18];
    const float* vis_b1  = (const float*)d_in[19];
    const float* vis_w2  = (const float*)d_in[20];
    const float* vis_b2  = (const float*)d_in[21];

    char* ws = (char*)d_ws;
    size_t off = 0;
    auto alloc = [&](size_t bytes) -> char* {
        char* p = ws + off;
        off += (bytes + 255) & ~(size_t)255;
        return p;
    };
    float*    coords = (float*)alloc((size_t)NROWS * 2 * sizeof(float));
    float*    h      = (float*)alloc((size_t)NROWS * HID * sizeof(float));
    _Float16* X      = (_Float16*)alloc((size_t)NROWS * KPAD * sizeof(_Float16));
    _Float16* z2     = (_Float16*)alloc((size_t)NROWS * HID * sizeof(_Float16));
    _Float16* g16    = (_Float16*)alloc((size_t)NROWS * 2 * HID * sizeof(_Float16));
    _Float16* w_in16 = (_Float16*)alloc((size_t)KPAD * HID * sizeof(_Float16));
    _Float16* w1_16  = (_Float16*)alloc((size_t)4 * HID * 2 * HID * sizeof(_Float16));
    _Float16* w2_16  = (_Float16*)alloc((size_t)4 * 2 * HID * HID * sizeof(_Float16));

    hipMemcpyAsync(coords, pos, (size_t)NROWS * 2 * sizeof(float),
                   hipMemcpyDeviceToDevice, stream);
    hipMemsetAsync(h, 0, (size_t)NROWS * HID * sizeof(float), stream);

    cvt_w_in_kernel<<<(KPAD * HID + 255) / 256, 256, 0, stream>>>(w_in, w_in16);
    cvt_f16_kernel<<<(4 * HID * 2 * HID + 255) / 256, 256, 0, stream>>>(mlp_w1, w1_16, 4 * HID * 2 * HID);
    cvt_f16_kernel<<<(4 * 2 * HID * HID + 255) / 256, 256, 0, stream>>>(mlp_w2, w2_16, 4 * 2 * HID * HID);

    for (int iter = 0; iter < 4; ++iter) {
        corr_assemble_kernel<<<NROWS, 64, 0, stream>>>(qf, fmaps, coords, h, X);
        // h = X @ W_in + b_in
        gemm_wmma_kernel<0><<<dim3(NROWS / 128, HID / 64), 256, 0, stream>>>(
            X, w_in16, b_in, nullptr, h, nullptr, KPAD, HID);
        for (int blk = 0; blk < 4; ++blk) {
            block_pre_kernel<<<QN, 256, 0, stream>>>(
                h, z2,
                ln1_g + blk * HID, ln1_b + blk * HID,
                conv_w + blk * HID * 5, conv_b + blk * HID,
                ln2_g + blk * HID, ln2_b + blk * HID);
            // g16 = gelu(z2 @ W1 + b1)   (f16 out)
            gemm_wmma_kernel<1><<<dim3(NROWS / 128, (2 * HID) / 64), 256, 0, stream>>>(
                z2, w1_16 + (size_t)blk * HID * 2 * HID, mlp_b1 + blk * 2 * HID,
                nullptr, nullptr, g16, HID, 2 * HID);
            // h = h + g16 @ W2 + b2
            gemm_wmma_kernel<2><<<dim3(NROWS / 128, HID / 64), 256, 0, stream>>>(
                g16, w2_16 + (size_t)blk * 2 * HID * HID, mlp_b2 + blk * HID,
                h, h, nullptr, 2 * HID, HID);
        }
        delta_update_kernel<<<NROWS / 256, 256, 0, stream>>>(coords, h, w_delta, b_delta);
    }
    finalize_kernel<<<NROWS, 64, 0, stream>>>(h, coords, vis_w1, vis_b1, vis_w2, vis_b2,
                                              (float*)d_out);
}
